// RecurrentNeuralNetworkDecoder_12506944766136
// MI455X (gfx1250) — matrix operand order repaired and checked by reading, not compiled, not run on hardware
//
#include <hip/hip_runtime.h>
#include <hip/hip_bf16.h>

// ---------------------------------------------------------------------------
// RNN decoder for MI455X (gfx1250): bf16 WMMA GEMMs, fp32 accumulate.
//   B=32, E=H=512, T=64, V=50257 (padded to 50272 for 16-wide WMMA tiles)
// Each GEMM wave owns one N-tile and computes BOTH M-tiles (M=32 -> 2 tiles),
// so every B (weight) tile is loaded once per wave instead of twice.
// ---------------------------------------------------------------------------

#define BATCH 32
#define HID   512
#define STEPS 64
#define VOCAB 50257
#define VPAD  50272              // 16*3142
#define NT1   3142               // N tiles of the big head

typedef __bf16 bf16_t;
typedef bf16_t v16bf __attribute__((ext_vector_type(16)));
typedef bf16_t v8bf  __attribute__((ext_vector_type(8)));
typedef float  v8f   __attribute__((ext_vector_type(8)));

__device__ __forceinline__ bf16_t f2bf(float f) {
    unsigned u = __builtin_bit_cast(unsigned, f);
    u += 0x7FFFu + ((u >> 16) & 1u);            // round-to-nearest-even
    unsigned short h = (unsigned short)(u >> 16);
    return __builtin_bit_cast(bf16_t, h);
}

// A-tile (16x32 bf16), row-major source, lda in elements.
// lane<16: row=M=lane,   K = {kb..kb+7, kb+16..kb+23}
// lane>=16: row=lane-16, K = {kb+8..kb+15, kb+24..kb+31}
__device__ __forceinline__ v16bf load_a_tile(const bf16_t* __restrict__ A, int lda,
                                             int mtile, int kb, int lane) {
    int r    = lane & 15;
    int half = lane >> 4;
    const bf16_t* base = A + (size_t)(mtile * 16 + r) * lda + kb + half * 8;
    v8bf lo = *(const v8bf*)(base);
    v8bf hi = *(const v8bf*)(base + 16);
    return __builtin_shufflevector(lo, hi, 0,1,2,3,4,5,6,7,8,9,10,11,12,13,14,15);
}

// B-tile (32x16 bf16): lane L holds column N=ntile*16+(L&15),
// K = {kb+8h..kb+8h+7, kb+16+8h..kb+16+8h+7} with h=L>>4 (same order as A).
__device__ __forceinline__ v16bf load_b_tile(const bf16_t* __restrict__ Bm, int ldb,
                                             int kb, int ntile, int lane) {
    const bf16_t* p = Bm + (size_t)(kb + (lane >> 4) * 8) * ldb + ntile * 16 + (lane & 15);
    v16bf v;
    #pragma unroll
    for (int i = 0; i < 8; ++i) {
        v[i]     = p[(size_t)i * ldb];
        v[i + 8] = p[(size_t)(i + 16) * ldb];
    }
    return v;
}

__device__ __forceinline__ v8f wmma_bf16(v16bf a, v16bf b, v8f c) {
    return __builtin_amdgcn_wmma_f32_16x16x32_bf16(false, a, false, b, (short)0, c,
                                                   false, false);
}

// ---------------------------------------------------------------------------
// Setup kernels
// ---------------------------------------------------------------------------
__global__ void cvt_bf16_kernel(const float* __restrict__ src, bf16_t* __restrict__ dst, int n) {
    int i = blockIdx.x * 256 + threadIdx.x;
    if (i < n) dst[i] = f2bf(src[i]);
}

__global__ void pad_v1_kernel(const float* __restrict__ V1, bf16_t* __restrict__ V1b) {
    int i = blockIdx.x * 256 + threadIdx.x;            // over 512*VPAD
    if (i >= HID * VPAD) return;
    int k = i / VPAD, n = i - k * VPAD;
    V1b[i] = (n < VOCAB) ? f2bf(V1[(size_t)k * VOCAB + n]) : f2bf(0.0f);
}

__global__ void pad_c1_kernel(const float* __restrict__ c1, float* __restrict__ c1p) {
    int i = blockIdx.x * 256 + threadIdx.x;
    if (i < VPAD) c1p[i] = (i < VOCAB) ? c1[i] : -1.0e30f;
}

__global__ void init_state_kernel(const int* __restrict__ tok, const float* __restrict__ emb,
                                  const float* __restrict__ h_prev,
                                  bf16_t* __restrict__ h0b, bf16_t* __restrict__ h1b,
                                  bf16_t* __restrict__ xcur) {
    int i = blockIdx.x * 256 + threadIdx.x;            // over 32*512
    if (i >= BATCH * HID) return;
    int b = i >> 9, e = i & 511;
    bf16_t hv = f2bf(h_prev[i]);
    h0b[i] = hv;
    h1b[i] = hv;
    xcur[i] = f2bf(emb[(size_t)tok[b] * HID + e]);
}

// ---------------------------------------------------------------------------
// Stage A/C: Hout = tanh(X*U + Hp*W + bias); optional fp32 side output.
// grid <<<4,256>>>: 32 waves -> 32 N-tiles, each wave does both M-tiles.
// ---------------------------------------------------------------------------
__global__ void rnn_cell_kernel(const bf16_t* __restrict__ X,  const bf16_t* __restrict__ U,
                                const bf16_t* __restrict__ Hp, const bf16_t* __restrict__ W,
                                const float*  __restrict__ bias,
                                bf16_t* __restrict__ Hout_bf,
                                float*  __restrict__ Hout_f32, int f32_row_stride) {
    int lane  = threadIdx.x & 31;
    int ntile = blockIdx.x * 8 + (threadIdx.x >> 5);   // 0..31
    v8f acc0 = {}, acc1 = {};
    #pragma unroll 4
    for (int k = 0; k < HID; k += 32) {
        v16bf bu = load_b_tile(U, HID, k, ntile, lane);
        acc0 = wmma_bf16(load_a_tile(X, HID, 0, k, lane), bu, acc0);
        acc1 = wmma_bf16(load_a_tile(X, HID, 1, k, lane), bu, acc1);
        v16bf bw = load_b_tile(W, HID, k, ntile, lane);
        acc0 = wmma_bf16(load_a_tile(Hp, HID, 0, k, lane), bw, acc0);
        acc1 = wmma_bf16(load_a_tile(Hp, HID, 1, k, lane), bw, acc1);
    }
    int col   = ntile * 16 + (lane & 15);
    int rbase = (lane >> 4) ? 8 : 0;
    float bv = bias[col];
    #pragma unroll
    for (int r = 0; r < 8; ++r) {
        float hv0 = tanhf(acc0[r] + bv);
        float hv1 = tanhf(acc1[r] + bv);
        int m0 = rbase + r, m1 = 16 + rbase + r;
        Hout_bf[(size_t)m0 * HID + col] = f2bf(hv0);
        Hout_bf[(size_t)m1 * HID + col] = f2bf(hv1);
        if (Hout_f32) {
            Hout_f32[(size_t)m0 * f32_row_stride + col] = hv0;
            Hout_f32[(size_t)m1 * f32_row_stride + col] = hv1;
        }
    }
}

// ---------------------------------------------------------------------------
// Stage B: x1 = tanh(softmax(h0*V0 + c0)).  Single workgroup (8 waves),
// logits staged in 64 KB LDS, per-row softmax.
// ---------------------------------------------------------------------------
__global__ void head0_kernel(const bf16_t* __restrict__ H0, const bf16_t* __restrict__ V0b,
                             const float* __restrict__ c0, bf16_t* __restrict__ X1out) {
    __shared__ float zbuf[BATCH * HID];                // 64 KB
    int lane = threadIdx.x & 31;
    int wave = threadIdx.x >> 5;
    for (int ntile = wave; ntile < 32; ntile += 8) {
        v8f acc0 = {}, acc1 = {};
        #pragma unroll 4
        for (int k = 0; k < HID; k += 32) {
            v16bf b = load_b_tile(V0b, HID, k, ntile, lane);
            acc0 = wmma_bf16(load_a_tile(H0, HID, 0, k, lane), b, acc0);
            acc1 = wmma_bf16(load_a_tile(H0, HID, 1, k, lane), b, acc1);
        }
        int col   = ntile * 16 + (lane & 15);
        int rbase = (lane >> 4) ? 8 : 0;
        float bv = c0[col];
        #pragma unroll
        for (int r = 0; r < 8; ++r) {
            zbuf[(rbase + r) * HID + col]      = acc0[r] + bv;
            zbuf[(16 + rbase + r) * HID + col] = acc1[r] + bv;
        }
    }
    __syncthreads();
    if (threadIdx.x < BATCH) {
        const float* z = zbuf + threadIdx.x * HID;
        float m = -3.4e38f;
        for (int j = 0; j < HID; ++j) m = fmaxf(m, z[j]);
        float s = 0.0f;
        for (int j = 0; j < HID; ++j) s += __expf(z[j] - m);
        float inv = 1.0f / s;
        bf16_t* xo = X1out + threadIdx.x * HID;
        for (int j = 0; j < HID; ++j) xo[j] = f2bf(tanhf(__expf(z[j] - m) * inv));
    }
}

// ---------------------------------------------------------------------------
// Stage D: Z = h1*V1 + c1 over padded vocab (fp32 logits into workspace).
// One wave per N-tile, both M-tiles -> V1 tile loaded exactly once.
// ---------------------------------------------------------------------------
__global__ void head1_gemm_kernel(const bf16_t* __restrict__ H1, const bf16_t* __restrict__ V1b,
                                  const float* __restrict__ c1p, float* __restrict__ Z) {
    int lane  = threadIdx.x & 31;
    int ntile = blockIdx.x * 8 + (threadIdx.x >> 5);
    if (ntile >= NT1) return;
    v8f acc0 = {}, acc1 = {};
    #pragma unroll 4
    for (int k = 0; k < HID; k += 32) {
        v16bf b = load_b_tile(V1b, VPAD, k, ntile, lane);
        acc0 = wmma_bf16(load_a_tile(H1, HID, 0, k, lane), b, acc0);
        acc1 = wmma_bf16(load_a_tile(H1, HID, 1, k, lane), b, acc1);
    }
    int col   = ntile * 16 + (lane & 15);
    int rbase = (lane >> 4) ? 8 : 0;
    float bv = c1p[col];
    #pragma unroll
    for (int r = 0; r < 8; ++r) {
        Z[(size_t)(rbase + r) * VPAD + col]      = acc0[r] + bv;
        Z[(size_t)(16 + rbase + r) * VPAD + col] = acc1[r] + bv;
    }
}

// ---------------------------------------------------------------------------
// Stage E: per-row max + argmax + sum(exp).  One block per batch row.
// ---------------------------------------------------------------------------
__global__ void softmax_reduce_kernel(const float* __restrict__ Z, float* __restrict__ rowmax,
                                      float* __restrict__ rowsum, int* __restrict__ rowarg) {
    __shared__ float smax[256];
    __shared__ int   sidx[256];
    __shared__ float ssum[256];
    int tid = threadIdx.x;
    const float* z = Z + (size_t)blockIdx.x * VPAD;
    float m = -3.4e38f; int mi = 0;
    for (int v = tid; v < VOCAB; v += 256) {
        float zv = z[v];
        if (zv > m) { m = zv; mi = v; }
    }
    smax[tid] = m; sidx[tid] = mi;
    __syncthreads();
    for (int s = 128; s > 0; s >>= 1) {
        if (tid < s) {
            if (smax[tid + s] > smax[tid] ||
                (smax[tid + s] == smax[tid] && sidx[tid + s] < sidx[tid])) {
                smax[tid] = smax[tid + s]; sidx[tid] = sidx[tid + s];
            }
        }
        __syncthreads();
    }
    float rm = smax[0];
    float sum = 0.0f;
    for (int v = tid; v < VOCAB; v += 256) sum += __expf(z[v] - rm);
    ssum[tid] = sum;
    __syncthreads();
    for (int s = 128; s > 0; s >>= 1) {
        if (tid < s) ssum[tid] += ssum[tid + s];
        __syncthreads();
    }
    if (tid == 0) {
        rowmax[blockIdx.x] = rm;
        rowsum[blockIdx.x] = ssum[0];
        rowarg[blockIdx.x] = sidx[0];
    }
}

// ---------------------------------------------------------------------------
// Stage F: y[b,t,:] = exp(z - max) / sum
// ---------------------------------------------------------------------------
__global__ void softmax_write_kernel(const float* __restrict__ Z, const float* __restrict__ rowmax,
                                     const float* __restrict__ rowsum,
                                     float* __restrict__ Yout, int t) {
    int i = blockIdx.x * 256 + threadIdx.x;            // over 32*VOCAB
    if (i >= BATCH * VOCAB) return;
    int b = i / VOCAB, v = i - b * VOCAB;
    float val = __expf(Z[(size_t)b * VPAD + v] - rowmax[b]) / rowsum[b];
    Yout[((size_t)b * STEPS + t) * VOCAB + v] = val;
}

// ---------------------------------------------------------------------------
// Stage G: greedy feedback — embed argmax token for next step.
// ---------------------------------------------------------------------------
__global__ void gather_next_kernel(const int* __restrict__ rowarg, const float* __restrict__ emb,
                                   bf16_t* __restrict__ xcur) {
    int i = blockIdx.x * 256 + threadIdx.x;            // over 32*512
    if (i >= BATCH * HID) return;
    int b = i >> 9, e = i & 511;
    xcur[i] = f2bf(emb[(size_t)rowarg[b] * HID + e]);
}

// ---------------------------------------------------------------------------
extern "C" void kernel_launch(void* const* d_in, const int* in_sizes, int n_in,
                              void* d_out, int out_size, void* d_ws, size_t ws_size,
                              hipStream_t stream) {
    const int*   x      = (const int*)d_in[0];
    const float* h_prev = (const float*)d_in[3];
    const float* emb    = (const float*)d_in[4];
    const float* U0 = (const float*)d_in[5];
    const float* W0 = (const float*)d_in[6];
    const float* b0 = (const float*)d_in[7];
    const float* V0 = (const float*)d_in[8];
    const float* c0 = (const float*)d_in[9];
    const float* U1 = (const float*)d_in[10];
    const float* W1 = (const float*)d_in[11];
    const float* b1 = (const float*)d_in[12];
    const float* V1 = (const float*)d_in[13];
    const float* c1 = (const float*)d_in[14];

    float* out_h = (float*)d_out;                       // (32, 64, 512)
    float* out_y = out_h + (size_t)BATCH * STEPS * HID; // (32, 64, 50257)

    // --- workspace bump allocator (256B aligned) ---
    char* wsb = (char*)d_ws;
    size_t off = 0;
    auto alloc = [&](size_t bytes) -> void* {
        void* p = wsb + off;
        off = (off + bytes + 255) & ~(size_t)255;
        return p;
    };
    const size_t SQ = (size_t)HID * HID;                 // 262144 elems
    bf16_t* U0b = (bf16_t*)alloc(SQ * 2);
    bf16_t* W0b = (bf16_t*)alloc(SQ * 2);
    bf16_t* V0b = (bf16_t*)alloc(SQ * 2);
    bf16_t* U1b = (bf16_t*)alloc(SQ * 2);
    bf16_t* W1b = (bf16_t*)alloc(SQ * 2);
    bf16_t* V1b = (bf16_t*)alloc((size_t)HID * VPAD * 2);
    float*  c1p = (float*)alloc((size_t)VPAD * 4);
    float*  Z   = (float*)alloc((size_t)BATCH * VPAD * 4);
    bf16_t* xcur = (bf16_t*)alloc((size_t)BATCH * HID * 2);
    bf16_t* x1b  = (bf16_t*)alloc((size_t)BATCH * HID * 2);
    bf16_t* h0b  = (bf16_t*)alloc((size_t)2 * BATCH * HID * 2);  // ping-pong
    bf16_t* h1b  = (bf16_t*)alloc((size_t)2 * BATCH * HID * 2);
    float*  rmax = (float*)alloc(BATCH * 4);
    float*  rsum = (float*)alloc(BATCH * 4);
    int*    rarg = (int*)alloc(BATCH * 4);
    (void)ws_size; (void)n_in; (void)in_sizes; (void)out_size;

    // --- one-time setup ---
    const int nsq_blk = (int)((SQ + 255) / 256);
    cvt_bf16_kernel<<<nsq_blk, 256, 0, stream>>>(U0, U0b, (int)SQ);
    cvt_bf16_kernel<<<nsq_blk, 256, 0, stream>>>(W0, W0b, (int)SQ);
    cvt_bf16_kernel<<<nsq_blk, 256, 0, stream>>>(V0, V0b, (int)SQ);
    cvt_bf16_kernel<<<nsq_blk, 256, 0, stream>>>(U1, U1b, (int)SQ);
    cvt_bf16_kernel<<<nsq_blk, 256, 0, stream>>>(W1, W1b, (int)SQ);
    pad_v1_kernel<<<(HID * VPAD + 255) / 256, 256, 0, stream>>>(V1, V1b);
    pad_c1_kernel<<<(VPAD + 255) / 256, 256, 0, stream>>>(c1, c1p);
    init_state_kernel<<<(BATCH * HID + 255) / 256, 256, 0, stream>>>(
        x, emb, h_prev, h0b, h1b, xcur);

    const size_t ST = (size_t)BATCH * HID;
    // --- 64 sequential decode steps ---
    for (int t = 0; t < STEPS; ++t) {
        bf16_t* h0p = h0b + (size_t)(t & 1) * ST;
        bf16_t* h0n = h0b + (size_t)((t & 1) ^ 1) * ST;
        bf16_t* h1p = h1b + (size_t)(t & 1) * ST;
        bf16_t* h1n = h1b + (size_t)((t & 1) ^ 1) * ST;

        // layer 0 cell
        rnn_cell_kernel<<<4, 256, 0, stream>>>(xcur, U0b, h0p, W0b, b0,
                                               h0n, (float*)nullptr, 0);
        // head 0 -> x1
        head0_kernel<<<1, 256, 0, stream>>>(h0n, V0b, c0, x1b);
        // layer 1 cell (h1 also goes to output h[:, t, :])
        rnn_cell_kernel<<<4, 256, 0, stream>>>(x1b, U1b, h1p, W1b, b1,
                                               h1n, out_h + (size_t)t * HID,
                                               STEPS * HID);
        // big vocab head
        head1_gemm_kernel<<<(NT1 + 7) / 8, 256, 0, stream>>>(h1n, V1b, c1p, Z);
        softmax_reduce_kernel<<<BATCH, 256, 0, stream>>>(Z, rmax, rsum, rarg);
        softmax_write_kernel<<<(BATCH * VOCAB + 255) / 256, 256, 0, stream>>>(
            Z, rmax, rsum, out_y, t);
        gather_next_kernel<<<(BATCH * HID + 255) / 256, 256, 0, stream>>>(rarg, emb, xcur);
    }
}
